// LSTM_NodeLinWeightsDistribution_74869869904304
// MI455X (gfx1250) — matrix operand order repaired and checked by reading, not compiled, not run on hardware
//
#include <hip/hip_runtime.h>

// CDNA5 WMMA vector types (wave32)
typedef __attribute__((ext_vector_type(16))) _Float16 v16h;
typedef __attribute__((ext_vector_type(8)))  float    v8f;

#define Bsz 2048
#define Tn  128
#define Hd  48
#define Ln  3
#define FH  192   // 4*H

// Dynamic LDS layout:
//  [0, 147456)        : f16 weights, 72 fragments x 1024 halfs
//                       frag(l,mat,nt) base = ((l*2+mat)*12+nt)*1024
//                       +0   : K-slice0 (K=0..31),  32 lanes x 16 halfs
//                       +512 : K-slice1 (K=32..63), 32 lanes x 16 halfs (K>=48 zero)
//  [147456, 150336)   : float params: bsum[576], Wi[48], bi[48], Wp[48]
//  [150336, 164672)   : f16 A-layout state tiles: xa[1024], ha[3][2][1024]
//  [164672, 170880)   : float: h32[768] (top-layer h), aemb[768], inp[16]
#define WLDS_BYTES  147456
#define PBUF_BYTES  2880
#define ATILE_OFF   (WLDS_BYTES + PBUF_BYTES)          // 150336 (32B aligned)
#define F32_OFF     (ATILE_OFF + 7 * 1024 * 2)         // 164672
#define SMEM_BYTES  (F32_OFF + (768 + 768 + 16) * 4)   // 170880

__device__ __forceinline__ float sigm(float x) { return 1.0f / (1.0f + __expf(-x)); }

// Position of element (m, k) inside an A-layout tile (2 K-slices x 32 lanes x 16 halfs).
// 16-bit A 16x32 layout: lanes 0-15: V0..3=K0..7, V4..7=K16..23; lanes 16-31: +8.
__device__ __forceinline__ int a_addr(int m, int j) {
  const int slice = j >> 5;          // 0: K 0..31, 1: K 32..47
  const int kb    = j & 31;          // for slice 1, kb in 0..15 -> V0..3
  const int hi8   = (kb >> 3) & 1;   // upper lane half holds K%16 >= 8
  const int v     = ((kb >> 4) << 2) + ((kb & 7) >> 1);
  const int p     = kb & 1;
  return slice * 512 + (m + 16 * hi8) * 16 + v * 2 + p;
}

__global__ __launch_bounds__(32) void lstm_dist_kernel(
    const float* __restrict__ adj, const float* __restrict__ eps,
    const float* __restrict__ Wi,  const float* __restrict__ bi,
    const float* __restrict__ Wa,  const float* __restrict__ ba,
    const float* __restrict__ Wp,  const float* __restrict__ bp,
    const float* __restrict__ Wih, const float* __restrict__ Whh,
    const float* __restrict__ bih, const float* __restrict__ bhh,
    const float* __restrict__ h0,  const float* __restrict__ c0,
    float* __restrict__ out) {
  extern __shared__ char smem[];
  _Float16* wlds = (_Float16*)smem;
  float* bsum = (float*)(smem + WLDS_BYTES);  // 576
  float* wi_s = bsum + 576;                   // 48
  float* bi_s = wi_s + 48;                    // 48
  float* wp_s = bi_s + 48;                    // 48
  _Float16* xa = (_Float16*)(smem + ATILE_OFF);  // 1024 halfs
  _Float16* ha = xa + 1024;                      // 6 x 1024 halfs
  float* h32  = (float*)(smem + F32_OFF);        // 768
  float* aemb = h32 + 768;                       // 768
  float* inp  = aemb + 768;                      // 16

  const int lane = threadIdx.x;      // one wave32 per workgroup
  const int b0   = blockIdx.x * 16;  // 16 batch rows per wave

  // ---- one-time: weights f32 -> f16, WMMA B-layout (zero-padded K-slice 1) ----
  // B 32x16 layout: lane l holds column n=l%16; lanes 0-15: K=0..7,16..23, lanes 16-31: K=8..15,24..31.
  for (int idx = lane; idx < 72 * 1024; idx += 32) {
    const int j  = idx & 15;
    const int ln = (idx >> 4) & 31;
    const int ks = (idx >> 9) & 1;
    const int f  = idx >> 10;
    const int nt = f % 12;
    const int fm = f / 12;  // l*2 + mat
    const int n  = nt * 16 + (ln & 15);
    const int k  = ks * 32 + ((j >> 3) << 4) + ((ln >> 4) << 3) + (j & 7);
    const float* src = (fm & 1) ? Whh : Wih;
    const float v = (k < Hd) ? src[((fm >> 1) * FH + n) * Hd + k] : 0.0f;
    wlds[idx] = (_Float16)v;
  }
  // small params
  for (int idx = lane; idx < 576; idx += 32) bsum[idx] = bih[idx] + bhh[idx];
  for (int idx = lane; idx < 48; idx += 32) {
    wi_s[idx] = Wi[idx];
    bi_s[idx] = bi[idx];
    wp_s[idx] = Wp[idx];
  }

  // ---- zero the A-layout state tiles (xa + 6 h tiles), then broadcast h0 ----
  {
    unsigned int* z = (unsigned int*)xa;
    for (int idx = lane; idx < 7 * 512; idx += 32) z[idx] = 0u;
  }
  for (int e = lane; e < 16 * Hd; e += 32) {
    const int m = e / Hd, j = e % Hd;
#pragma unroll
    for (int l3 = 0; l3 < Ln; ++l3)
      ha[(l3 * 2 + 0) * 1024 + a_addr(m, j)] = (_Float16)h0[l3 * Hd + j];
  }
  if (lane < 16) inp[lane] = 0.0f;

  // adj_emb = adj @ Wa.T + ba  (one-time, [16][48] f32)
  for (int e = lane; e < 16 * Hd; e += 32) {
    const int m = e / Hd, j = e % Hd;
    const float* arow = adj + (size_t)(b0 + m) * Tn;
    const float* wrow = Wa + (size_t)j * Tn;
    float acc = ba[j];
    for (int t = 0; t < Tn; ++t) acc += arow[t] * wrow[t];
    aemb[e] = acc;
  }

  // ---- per-lane register-resident cell state: creg[l][hc][v] ----
  const int n  = lane & 15;
  const int hi = (lane & 16) ? 8 : 0;
  float creg[Ln][3][8];
#pragma unroll
  for (int l3 = 0; l3 < Ln; ++l3)
#pragma unroll
    for (int hc = 0; hc < 3; ++hc) {
      const float cv = c0[l3 * Hd + hc * 16 + n];
#pragma unroll
      for (int v = 0; v < 8; ++v) creg[l3][hc][v] = cv;
    }
  __syncthreads();

  // ---- the 128-step recurrence, fully on-chip ----
  int s = 0;  // h read-slot
  for (int t = 0; t < Tn; ++t) {
    // x = (inp * Wi^T + bi) * adj_emb, written straight into f16 A-layout
    for (int e = lane; e < 16 * Hd; e += 32) {
      const int m = e / Hd, j = e % Hd;
      xa[a_addr(m, j)] = (_Float16)((inp[m] * wi_s[j] + bi_s[j]) * aemb[e]);
    }

#pragma unroll
    for (int l3 = 0; l3 < Ln; ++l3) {
      const _Float16* xat = (l3 == 0) ? xa : (ha + ((l3 - 1) * 2 + (1 - s)) * 1024);
      const _Float16* hat = ha + (l3 * 2 + s) * 1024;
      _Float16* haw = ha + (l3 * 2 + (1 - s)) * 1024;

      const v16h ax0 = *(const v16h*)(xat);
      const v16h ax1 = *(const v16h*)(xat + 512 + lane * 0) /*dummy*/,
                 ax1r = *(const v16h*)(xat + 512 + lane * 16);
      (void)ax1;
      const v16h ax0r = *(const v16h*)(xat + lane * 16);
      const v16h ah0 = *(const v16h*)(hat + lane * 16);
      const v16h ah1 = *(const v16h*)(hat + 512 + lane * 16);

#pragma unroll
      for (int hc = 0; hc < 3; ++hc) {
        v8f acc[4];
#pragma unroll
        for (int g = 0; g < 4; ++g) {  // i, f, g, o
          const int nt = g * 3 + hc;
          const _Float16* fih = wlds + (size_t)((l3 * 2 + 0) * 12 + nt) * 1024;
          const _Float16* fhh = wlds + (size_t)((l3 * 2 + 1) * 12 + nt) * 1024;
          v8f a = {};  // zero C operand; bias added in epilogue
          a = __builtin_amdgcn_wmma_f32_16x16x32_f16(
              false, ax0r, false, *(const v16h*)(fih + lane * 16), (short)0, a, false, false);
          a = __builtin_amdgcn_wmma_f32_16x16x32_f16(
              false, ax1r, false, *(const v16h*)(fih + 512 + lane * 16), (short)0, a, false, false);
          a = __builtin_amdgcn_wmma_f32_16x16x32_f16(
              false, ah0, false, *(const v16h*)(fhh + lane * 16), (short)0, a, false, false);
          a = __builtin_amdgcn_wmma_f32_16x16x32_f16(
              false, ah1, false, *(const v16h*)(fhh + 512 + lane * 16), (short)0, a, false, false);
          acc[g] = a;
        }
        const float bvi = bsum[l3 * FH + 0 * Hd + hc * 16 + n];
        const float bvf = bsum[l3 * FH + 1 * Hd + hc * 16 + n];
        const float bvg = bsum[l3 * FH + 2 * Hd + hc * 16 + n];
        const float bvo = bsum[l3 * FH + 3 * Hd + hc * 16 + n];
        const int j = hc * 16 + n;
#pragma unroll
        for (int v = 0; v < 8; ++v) {
          const int m = v + hi;
          const float iv = sigm(acc[0][v] + bvi);
          const float fv = sigm(acc[1][v] + bvf);
          const float gv = tanhf(acc[2][v] + bvg);
          const float ov = sigm(acc[3][v] + bvo);
          const float cn = fv * creg[l3][hc][v] + iv * gv;
          creg[l3][hc][v] = cn;
          const float hn = ov * tanhf(cn);
          haw[a_addr(m, j)] = (_Float16)hn;   // f16 A-layout for next matmuls
          if (l3 == Ln - 1) h32[m * Hd + j] = hn;  // f32 copy for projection
        }
      }
    }

    // projection + reparameterized masked sample
    if (lane < 16) {
      float acc = bp[0];
#pragma unroll
      for (int j = 0; j < Hd; ++j) acc += h32[lane * Hd + j] * wp_s[j];
      const size_t gi = (size_t)(b0 + lane) * Tn + t;
      const float sample = (acc + eps[gi]) * adj[gi];
      out[gi]   = sample;
      inp[lane] = sample;
    }
    __syncthreads();  // single-wave WG: nop, keeps LDS ordering explicit
    s ^= 1;
  }
}

extern "C" void kernel_launch(void* const* d_in, const int* in_sizes, int n_in,
                              void* d_out, int out_size, void* d_ws, size_t ws_size,
                              hipStream_t stream) {
  (void)in_sizes; (void)n_in; (void)d_ws; (void)ws_size; (void)out_size;
  const float* adj = (const float*)d_in[0];
  const float* eps = (const float*)d_in[1];
  const float* Wi  = (const float*)d_in[2];
  const float* bi  = (const float*)d_in[3];
  const float* Wa  = (const float*)d_in[4];
  const float* ba  = (const float*)d_in[5];
  const float* Wp  = (const float*)d_in[6];
  const float* bp  = (const float*)d_in[7];
  const float* Wih = (const float*)d_in[8];
  const float* Whh = (const float*)d_in[9];
  const float* bih = (const float*)d_in[10];
  const float* bhh = (const float*)d_in[11];
  const float* h0  = (const float*)d_in[12];
  const float* c0  = (const float*)d_in[13];
  float* out = (float*)d_out;

  dim3 grid(Bsz / 16);  // 128 independent 16-row recurrences
  dim3 block(32);       // one wave32 per workgroup
  lstm_dist_kernel<<<grid, block, SMEM_BYTES, stream>>>(
      adj, eps, Wi, bi, Wa, ba, Wp, bp, Wih, Whh, bih, bhh, h0, c0, out);
}
